// DIFFormerConv_10711648436656
// MI455X (gfx1250) — matrix operand order repaired and checked by reading, not compiled, not run on hardware
//
#include <hip/hip_runtime.h>
#include <hip/hip_bf16.h>
#include <math.h>

// ---------------- problem constants (from reference) ----------------
#define NNODES 100000
#define NEDGES 800000
#define DIN 64
#define DOUT 64
#define NH 4
#define HD 256            // NH * DOUT
#define KVP 128           // partial blocks for k^T v reduction
#define CHUNK 16          // nodes staged in LDS per iteration in kv_partial

static constexpr float GWEIGHT = 0.5f;
static constexpr float ALPHA_C = 0.5f;
static constexpr float LN_EPS_C = 1e-5f;

typedef __attribute__((ext_vector_type(16))) _Float16 v16h;
typedef __attribute__((ext_vector_type(8)))  float    v8f;
typedef int v4i_ __attribute__((vector_size(16)));
typedef __attribute__((address_space(1))) v4i_* gvec_p;   // global b128 payload
typedef __attribute__((address_space(3))) v4i_* lvec_p;   // LDS b128 payload

// ---------------- CDNA5 async global->LDS helpers ----------------
#if defined(__has_builtin)
#if __has_builtin(__builtin_amdgcn_global_load_async_to_lds_b128)
#define HAVE_ASYNC_LDS 1
#endif
#endif

__device__ __forceinline__ void g2l_async16(const float* g, float* l)
{
#ifdef HAVE_ASYNC_LDS
    __builtin_amdgcn_global_load_async_to_lds_b128((gvec_p)g, (lvec_p)l, 0, 0);
#else
    *(float4*)l = *(const float4*)g;
#endif
}

__device__ __forceinline__ void async_wait0()
{
#ifdef HAVE_ASYNC_LDS
#if __has_builtin(__builtin_amdgcn_s_wait_asynccnt)
    __builtin_amdgcn_s_wait_asynccnt(0);
#else
    asm volatile("s_wait_asynccnt 0x0" ::: "memory");
#endif
#endif
}

// =====================================================================
// Kernel 0: pack W_{q,k,v} into f16 B-fragments (16-bit B 32x16, contiguous-k
// layout): Bp[((ct*2+s)*32+lane)*16+e] = W[(s*32+16*hi+e)*256 + colbase+l]
// 96 KB total, L2-resident; turns each B fragment into one 32B load.
// =====================================================================
__global__ void pack_b(const float* __restrict__ Wq, const float* __restrict__ Wk,
                       const float* __restrict__ Wv, _Float16* __restrict__ Bp)
{
    const int t = blockIdx.x * blockDim.x + threadIdx.x;
    if (t >= 48 * 2 * 32 * 16) return;
    const int e    = t & 15;
    const int lane = (t >> 4) & 31;
    const int s    = (t >> 9) & 1;
    const int ct   = t >> 10;            // 0..47
    const int mat  = ct >> 4;
    const int l = lane & 15, hi = lane >> 4;
    const float* __restrict__ W = (mat == 0) ? Wq : (mat == 1) ? Wk : Wv;
    const int K   = s * 32 + 16 * hi + e;
    const int col = (ct & 15) * 16 + l;
    Bp[t] = (_Float16)W[(size_t)K * HD + col];
}

// =====================================================================
// Kernel 1: q/k/v = x @ W* + b* via V_WMMA_F32_16X16X32_F16 (wave32).
// Block = 8 waves; block owns a 16-node tile; wave does 6 of 48 col-tiles.
// Also accumulates sum(q^2), sum(k^2) for the global Frobenius norms.
// =====================================================================
__global__ __launch_bounds__(256) void qkv_wmma(
    const float* __restrict__ x, const _Float16* __restrict__ Bp,
    const float* __restrict__ bq, const float* __restrict__ bk,
    const float* __restrict__ bv,
    float* __restrict__ qo, float* __restrict__ ko, float* __restrict__ vo,
    float* __restrict__ sumsq)
{
    __shared__ float red[8][2];
    const int wave = threadIdx.x >> 5;
    const int lane = threadIdx.x & 31;
    const int l    = lane & 15;
    const int hi   = lane >> 4;
    const int node0 = blockIdx.x * 16;

    // A fragment: x tile [16 x 64], 16-bit A 16x32, contiguous k (16*hi + e)
    const float* __restrict__ xrow = x + (size_t)(node0 + l) * DIN;
    v16h afrag[2];
#pragma unroll
    for (int s = 0; s < 2; ++s) {
        const int kb = s * 32;
#pragma unroll
        for (int j = 0; j < 8; ++j) {
            afrag[s][j]     = (_Float16)xrow[kb + 16 * hi + j];
            afrag[s][8 + j] = (_Float16)xrow[kb + 16 * hi + 8 + j];
        }
    }

    float ssq = 0.f, ssk = 0.f;
#pragma unroll
    for (int t = 0; t < 6; ++t) {
        const int ct  = wave * 6 + t;      // 0..47
        const int mat = ct >> 4;           // 0=q 1=k 2=v
        const int c0  = (ct & 15) * 16;
        const float* __restrict__ bias = (mat == 0) ? bq : (mat == 1) ? bk : bv;
        float* __restrict__ out        = (mat == 0) ? qo : (mat == 1) ? ko : vo;

        v8f acc = {};
#pragma unroll
        for (int s = 0; s < 2; ++s) {
            const v16h bfrag = *(const v16h*)(Bp + ((size_t)(ct * 2 + s) * 32 + lane) * 16);
            acc = __builtin_amdgcn_wmma_f32_16x16x32_f16(
                false, afrag[s], false, bfrag, (short)0, acc, false, false);
        }
        const float bcol = bias[c0 + l];
#pragma unroll
        for (int i = 0; i < 8; ++i) {
            const float val = acc[i] + bcol;       // D: row i+8*hi, col c0+l
            out[(size_t)(node0 + i + 8 * hi) * HD + c0 + l] = val;
            if (mat == 0)      ssq += val * val;
            else if (mat == 1) ssk += val * val;
        }
    }
    for (int off = 16; off >= 1; off >>= 1) {
        ssq += __shfl_xor(ssq, off, 32);
        ssk += __shfl_xor(ssk, off, 32);
    }
    if (lane == 0) { red[wave][0] = ssq; red[wave][1] = ssk; }
    __syncthreads();
    if (threadIdx.x < 2) {
        float s = 0.f;
        for (int w = 0; w < 8; ++w) s += red[w][threadIdx.x];
        atomicAdd(&sumsq[threadIdx.x], s);
    }
}

// =====================================================================
// Kernel 2: vbar[n,d] = mean_h v[n,h,d]. GCN is linear and head-uniform,
// so head-mean commutes with the conv: 4x less gather/atomic traffic and
// vbar (25.6 MB) stays resident in the 192 MB L2.
// =====================================================================
__global__ void vbar_kernel(const float* __restrict__ v, float* __restrict__ vbar)
{
    const int t = blockIdx.x * blockDim.x + threadIdx.x;
    if (t >= NNODES * DOUT) return;
    const int n = t >> 6, d = t & 63;
    const float* vr = v + (size_t)n * HD;
    vbar[t] = 0.25f * (vr[d] + vr[64 + d] + vr[128 + d] + vr[192 + d]);
}

// =====================================================================
// Kernel 3: per-block partials of kv[h,m,d]=sum_n k[n,h,m]v[n,h,d],
// ks[h,m]=sum_n k, vs[h,d]=sum_n v. Chunks staged via async global->LDS.
// =====================================================================
__global__ __launch_bounds__(256) void kv_partial(
    const float* __restrict__ k, const float* __restrict__ v,
    float* __restrict__ kvp, float* __restrict__ ksp, float* __restrict__ vsp)
{
    __shared__ float sh[CHUNK * 512];   // 32 KB: [node][k row 256 | v row 256]
    const int t = threadIdx.x;
    const int h = t >> 6;
    const int b = blockIdx.x;
    const int per = (NNODES + KVP - 1) / KVP;
    const int n0 = b * per;
    const int n1 = (n0 + per < NNODES) ? (n0 + per) : NNODES;

    float acc[64];
#pragma unroll
    for (int d = 0; d < 64; ++d) acc[d] = 0.f;
    float aks = 0.f, avs = 0.f;

    for (int c = n0; c < n1; c += CHUNK) {
        __syncthreads();
        // stage CHUNK*(k row | v row) as b128s; rows >= n1 left stale but
        // never read (inner loop bounded by lim).
        for (int i4 = t; i4 < CHUNK * 128; i4 += 256) {
            const int nn = c + (i4 >> 7);
            const int j4 = i4 & 127;
            if (nn < n1) {
                const float* src = (j4 < 64) ? (k + (size_t)nn * HD + j4 * 4)
                                             : (v + (size_t)nn * HD + (j4 - 64) * 4);
                g2l_async16(src, &sh[i4 * 4]);
            }
        }
        async_wait0();
        __syncthreads();
        const int lim = ((n1 - c) < CHUNK) ? (n1 - c) : CHUNK;
        for (int n = 0; n < lim; ++n) {
            const float sk = sh[n * 512 + t];
            aks += sk;
            avs += sh[n * 512 + 256 + t];
            const float* vrow = &sh[n * 512 + 256 + h * 64];
#pragma unroll
            for (int d = 0; d < 64; ++d) acc[d] += sk * vrow[d];
        }
    }
    float* o = kvp + (size_t)b * 16384 + (size_t)t * 64;
#pragma unroll
    for (int d = 0; d < 64; ++d) o[d] = acc[d];
    ksp[b * 256 + t] = aks;
    vsp[b * 256 + t] = avs;
}

// =====================================================================
// Kernel 4: reduce partials, fold in 1/||k||_F, and emit kvs directly as
// f16 WMMA B-fragments (32 KB) for the attention-readout GEMM.
// =====================================================================
__global__ void kv_reduce(const float* __restrict__ kvp, const float* __restrict__ ksp,
                          const float* __restrict__ vsp, const float* __restrict__ sumsq,
                          _Float16* __restrict__ kvsh, float* __restrict__ kss,
                          float* __restrict__ vsum)
{
    const int j = blockIdx.x * blockDim.x + threadIdx.x;
    const float rk = 1.0f / sqrtf(sumsq[1]);
    if (j < 16384) {
        float s = 0.f;
        for (int p = 0; p < KVP; ++p) s += kvp[(size_t)p * 16384 + j];
        const float val = s * rk;
        // j = h*4096 + m*64 + d  ->  B-fragment (h, ct=d/16, s=m/32) coords
        const int h = j >> 12, rem = j & 4095;
        const int m = rem >> 6, d = rem & 63;
        const int sb = m >> 5, hi = (m >> 4) & 1, e = m & 15;
        const int ct = d >> 4, l = d & 15;
        kvsh[(((size_t)(h * 4 + ct) * 2 + sb) * 32 + hi * 16 + l) * 16 + e] = (_Float16)val;
    }
    if (j < 256) {
        float s1 = 0.f, s2 = 0.f;
        for (int p = 0; p < KVP; ++p) { s1 += ksp[p * 256 + j]; s2 += vsp[p * 256 + j]; }
        kss[j]  = s1 * rk;
        vsum[j] = s2;
    }
}

// =====================================================================
// Kernel 5: col-degree via atomics.
// =====================================================================
__global__ void deg_kernel(const int* __restrict__ ei, float* __restrict__ deg)
{
    const int e = blockIdx.x * blockDim.x + threadIdx.x;
    if (e < NEDGES) atomicAdd(&deg[ei[NEDGES + e]], 1.0f);
}

// =====================================================================
// Kernel 6: GCN scatter on head-averaged vbar. 16 lanes/edge, float4.
// =====================================================================
__global__ __launch_bounds__(256) void gcn_scatter(
    const int* __restrict__ ei, const float* __restrict__ deg,
    const float* __restrict__ vbar, float* __restrict__ gcn)
{
    const int t = blockIdx.x * blockDim.x + threadIdx.x;
    const int e = t >> 4;
    if (e >= NEDGES) return;
    const int d0  = (t & 15) * 4;
    const int row = ei[e];
    const int col = ei[NEDGES + e];
    __builtin_prefetch(vbar + (size_t)row * 64 + d0, 0, 3);   // global_prefetch_b8
    const float dr = deg[row];
    const float w  = (dr > 0.f) ? (1.0f / sqrtf(dr)) * (1.0f / sqrtf(deg[col])) : 0.f;
    const float4 vb = *(const float4*)(vbar + (size_t)row * 64 + d0);
    float* g = gcn + (size_t)col * 64 + d0;
    atomicAdd(g + 0, w * vb.x);
    atomicAdd(g + 1, w * vb.y);
    atomicAdd(g + 2, w * vb.z);
    atomicAdd(g + 3, w * vb.w);
}

// =====================================================================
// Kernel 7: attention readout as WMMA + mix + residual + LayerNorm.
// Block = 32 nodes, 8 waves; wave w owns (subtile st=w/4, head h=w&3) and
// computes num[16x64] = q_tile @ kvs[h] on the matrix pipe. Per-head
// results go to LDS (aliasing the q staging buffer), then a wave-per-node
// pass does head-mean, GCN mix, residuals and a wave32-shuffle LayerNorm.
// =====================================================================
__global__ __launch_bounds__(256) void final_wmma(
    const float* __restrict__ q,   const _Float16* __restrict__ kvsh,
    const float* __restrict__ kss, const float* __restrict__ vsum,
    const float* __restrict__ gcn, const float* __restrict__ sumsq,
    const float* __restrict__ x,   const float* __restrict__ xo,
    const float* __restrict__ lng, const float* __restrict__ lnb,
    float* __restrict__ out)
{
    __shared__ float smem[32 * 256];   // phase1: q rows; phase2: att[4][32][64]
    __shared__ float kssh[256];
    __shared__ float vssh[256];
    __shared__ float dnm[8][16];
    const int tid = threadIdx.x;
    const int n0  = blockIdx.x * 32;

    // ---- stage q[n0..n0+32) via async global->LDS ----
    for (int i4 = tid; i4 < 32 * 64; i4 += 256)
        g2l_async16(q + (size_t)n0 * HD + i4 * 4, &smem[i4 * 4]);
    async_wait0();
    kssh[tid] = kss[tid];
    vssh[tid] = vsum[tid];
    __syncthreads();

    const int wave = tid >> 5;
    const int lane = tid & 31;
    const int l    = lane & 15;
    const int hi   = lane >> 4;
    const int st   = wave >> 2;        // subtile 0/1
    const int h    = wave & 3;         // head
    const float rq = 1.0f / sqrtf(sumsq[0]);

    // ---- A fragment: q subtile [16 x 64] (K = head feature m) ----
    v16h afrag[2];
#pragma unroll
    for (int s = 0; s < 2; ++s) {
        const int base = (st * 16 + l) * 256 + h * 64 + s * 32;
#pragma unroll
        for (int j = 0; j < 8; ++j) {
            afrag[s][j]     = (_Float16)smem[base + 16 * hi + j];
            afrag[s][8 + j] = (_Float16)smem[base + 16 * hi + 8 + j];
        }
    }
    // ---- denominators: dot(q[n], kss[h]) (redundant across hi halves) ----
    {
        float dsum = 0.f;
        const int base = (st * 16 + l) * 256 + h * 64;
#pragma unroll
        for (int m = 0; m < 64; ++m) dsum += smem[base + m] * kssh[h * 64 + m];
        if (hi == 0) dnm[wave][l] = dsum * rq + (float)NNODES;
    }
    __syncthreads();   // everyone done reading smem-as-q; reuse as att

    float* att = smem;                 // att[h][32][64], 4*2048 floats
#pragma unroll
    for (int ct = 0; ct < 4; ++ct) {
        const int c0 = ct * 16;
        v8f acc = {};
#pragma unroll
        for (int s = 0; s < 2; ++s) {
            const v16h bfrag =
                *(const v16h*)(kvsh + (((size_t)(h * 4 + ct) * 2 + s) * 32 + lane) * 16);
            acc = __builtin_amdgcn_wmma_f32_16x16x32_f16(
                false, afrag[s], false, bfrag, (short)0, acc, false, false);
        }
#pragma unroll
        for (int i = 0; i < 8; ++i) {
            const int r = i + 8 * hi;                       // row in subtile
            const float num = acc[i] * rq + vssh[h * 64 + c0 + l];
            att[h * 2048 + (st * 16 + r) * 64 + c0 + l] = num / dnm[wave][r];
        }
    }
    __syncthreads();

    // ---- head mix + GCN mix + residual + LayerNorm (wave per node) ----
    for (int g = 0; g < 4; ++g) {
        const int nib = wave + 8 * g;          // node-in-block
        const int n   = n0 + nib;
        float res[2];
#pragma unroll
        for (int p = 0; p < 2; ++p) {
            const int d = lane + 32 * p;
            const float asum = att[0 * 2048 + nib * 64 + d] + att[1 * 2048 + nib * 64 + d]
                             + att[2 * 2048 + nib * 64 + d] + att[3 * 2048 + nib * 64 + d];
            float pre = (1.f - GWEIGHT) * (asum * 0.25f) + GWEIGHT * gcn[(size_t)n * 64 + d];
            pre += xo[(size_t)n * 64 + d];
            res[p] = pre * ALPHA_C + x[(size_t)n * 64 + d] * (1.f - ALPHA_C);
        }
        float s  = res[0] + res[1];
        float ss = res[0] * res[0] + res[1] * res[1];
        for (int off = 16; off >= 1; off >>= 1) {
            s  += __shfl_xor(s, off, 32);
            ss += __shfl_xor(ss, off, 32);
        }
        const float mu  = s * (1.f / 64.f);
        const float var = ss * (1.f / 64.f) - mu * mu;
        const float inv = 1.0f / sqrtf(var + LN_EPS_C);
#pragma unroll
        for (int p = 0; p < 2; ++p) {
            const int d = lane + 32 * p;
            out[(size_t)n * 64 + d] = (res[p] - mu) * inv * lng[d] + lnb[d];
        }
    }
}

// =====================================================================
extern "C" void kernel_launch(void* const* d_in, const int* in_sizes, int n_in,
                              void* d_out, int out_size, void* d_ws, size_t ws_size,
                              hipStream_t stream)
{
    const float* x   = (const float*)d_in[0];
    const float* xo  = (const float*)d_in[1];
    const float* Wq  = (const float*)d_in[2];
    const float* bq  = (const float*)d_in[3];
    const float* Wk  = (const float*)d_in[4];
    const float* bk  = (const float*)d_in[5];
    const float* Wv  = (const float*)d_in[6];
    const float* bv  = (const float*)d_in[7];
    const float* lng = (const float*)d_in[8];
    const float* lnb = (const float*)d_in[9];
    const int*   ei  = (const int*)d_in[10];
    float* outp = (float*)d_out;

    // ---- workspace layout: f16 regions first (32B aligned), then f32 ----
    _Float16* bp   = (_Float16*)d_ws;               // 48*2*32*16 = 49152 halves
    _Float16* kvsh = bp + 49152;                    // 16384 halves
    float* fws  = (float*)(kvsh + 16384);           // float region (128 KB offset)
    float* qbuf = fws;                              // N*256
    float* kbuf = qbuf + (size_t)NNODES * HD;
    float* vbuf = kbuf + (size_t)NNODES * HD;
    float* vbar = vbuf + (size_t)NNODES * HD;       // N*64
    float* gcn  = vbar + (size_t)NNODES * DOUT;     // N*64
    float* deg  = gcn  + (size_t)NNODES * DOUT;     // N
    float* kvp  = deg  + NNODES;                    // KVP*16384
    float* ksp  = kvp  + (size_t)KVP * 16384;       // KVP*256
    float* vsp  = ksp  + KVP * 256;                 // KVP*256
    float* kss  = vsp  + KVP * 256;                 // 256
    float* vsum = kss  + 256;                       // 256
    float* sumsq = vsum + 256;                      // 2

    (void)hipMemsetAsync(deg,   0, NNODES * sizeof(float), stream);
    (void)hipMemsetAsync(gcn,   0, (size_t)NNODES * DOUT * sizeof(float), stream);
    (void)hipMemsetAsync(sumsq, 0, 2 * sizeof(float), stream);

    pack_b<<<(48 * 2 * 32 * 16 + 255) / 256, 256, 0, stream>>>(Wq, Wk, Wv, bp);
    qkv_wmma<<<NNODES / 16, 256, 0, stream>>>(x, bp, bq, bk, bv,
                                              qbuf, kbuf, vbuf, sumsq);
    vbar_kernel<<<(NNODES * DOUT + 255) / 256, 256, 0, stream>>>(vbuf, vbar);
    kv_partial<<<KVP, 256, 0, stream>>>(kbuf, vbuf, kvp, ksp, vsp);
    kv_reduce<<<64, 256, 0, stream>>>(kvp, ksp, vsp, sumsq, kvsh, kss, vsum);
    deg_kernel<<<(NEDGES + 255) / 256, 256, 0, stream>>>(ei, deg);
    gcn_scatter<<<(NEDGES * 16) / 256, 256, 0, stream>>>(ei, deg, vbar, gcn);
    final_wmma<<<NNODES / 32, 256, 0, stream>>>(qbuf, kvsh, kss, vsum, gcn, sumsq,
                                                x, xo, lng, lnb, outp);
}
